// Matching_68753836474835
// MI455X (gfx1250) — matrix operand order repaired and checked by reading, not compiled, not run on hardware
//
#include <hip/hip_runtime.h>
#include <hip/hip_bf16.h>
#include <cstddef>

typedef __attribute__((ext_vector_type(16))) _Float16 v16h;
typedef __attribute__((ext_vector_type(8)))  _Float16 v8h;
typedef __attribute__((ext_vector_type(8)))  float    v8f;

#define BN 4
#define CF 96
#define HH 192
#define WW 192

// ---------------------------------------------------------------------------
// 1) upflow: transposed conv, (B,2,96,96) -> (B,2,192,192), 4x4 kernel, g=2
//    out[y,x] = sum_{i,j, (y+i-2) even, (x+j-2) even} in[(y+i-2)/2,(x+j-2)/2]
//               * w[c,0,3-i,3-j]
// ---------------------------------------------------------------------------
__global__ void upflow_kernel(const float* __restrict__ fin,
                              const float* __restrict__ w,
                              float* __restrict__ fout)
{
    int idx = blockIdx.x * blockDim.x + threadIdx.x;
    const int n = BN * 2 * HH * WW;
    if (idx >= n) return;
    int x = idx % WW;
    int y = (idx / WW) % HH;
    int c = (idx / (WW * HH)) % 2;
    int b = idx / (WW * HH * 2);
    const int Hi = HH / 2, Wi = WW / 2;
    float acc = 0.f;
    #pragma unroll
    for (int i = 0; i < 4; ++i) {
        int p = y + i - 2;
        if (p < 0 || p >= 2 * Hi || (p & 1)) continue;
        int py = p >> 1;
        #pragma unroll
        for (int j = 0; j < 4; ++j) {
            int q = x + j - 2;
            if (q < 0 || q >= 2 * Wi || (q & 1)) continue;
            int qx = q >> 1;
            acc += fin[((size_t)(b * 2 + c) * Hi + py) * Wi + qx] *
                   w[(c * 4 + (3 - i)) * 4 + (3 - j)];
        }
    }
    fout[idx] = acc;
}

// ---------------------------------------------------------------------------
// 2) NCHW f32 -> NHWC f16 layout transform (for features)
// ---------------------------------------------------------------------------
__global__ void nchw_to_nhwc_f16(const float* __restrict__ in,
                                 _Float16* __restrict__ out)
{
    size_t idx = (size_t)blockIdx.x * blockDim.x + threadIdx.x;
    const size_t n = (size_t)BN * CF * HH * WW;
    if (idx >= n) return;
    int c = (int)(idx % CF);
    size_t p = idx / CF;
    int x = (int)(p % WW);
    int y = (int)((p / WW) % HH);
    int b = (int)(p / ((size_t)WW * HH));
    out[idx] = (_Float16)in[((size_t)(b * CF + c) * HH + y) * WW + x];
}

// ---------------------------------------------------------------------------
// 3) backwarp: bilinear warp NHWC f16 features by flow*2.5 (zero outside)
//    one block per pixel, threads over channels (C=96, coalesced)
// ---------------------------------------------------------------------------
__global__ void backwarp_kernel(const _Float16* __restrict__ f2,   // NHWC
                                const float* __restrict__ flow,    // NCHW (B,2,H,W)
                                _Float16* __restrict__ out)        // NHWC
{
    int pix = blockIdx.x;
    int x = pix % WW;
    int y = (pix / WW) % HH;
    int b = pix / (WW * HH);
    int c = threadIdx.x;
    float fx = flow[((size_t)(b * 2 + 0) * HH + y) * WW + x] * 2.5f;
    float fy = flow[((size_t)(b * 2 + 1) * HH + y) * WW + x] * 2.5f;
    float gx = (float)x + fx;
    float gy = (float)y + fy;
    float x0f = floorf(gx), y0f = floorf(gy);
    float wx = gx - x0f, wy = gy - y0f;
    int x0 = (int)x0f, y0 = (int)y0f;
    float acc = 0.f;
    #pragma unroll
    for (int dy = 0; dy < 2; ++dy) {
        #pragma unroll
        for (int dx = 0; dx < 2; ++dx) {
            int xi = x0 + dx, yi = y0 + dy;
            float wgt = (dy ? wy : 1.f - wy) * (dx ? wx : 1.f - wx);
            if (xi >= 0 && xi < WW && yi >= 0 && yi < HH)
                acc += wgt * (float)f2[((size_t)(b * HH + yi) * WW + xi) * CF + c];
        }
    }
    out[((size_t)(b * HH + y) * WW + x) * CF + c] = (_Float16)acc;
}

// ---------------------------------------------------------------------------
// 4) correlation: 7x7 window, mean over C=96, leaky, NHWC f16 padded to 64ch
//    one block (64 threads) per pixel; f1 vector staged in LDS
// ---------------------------------------------------------------------------
__global__ void corr_kernel(const _Float16* __restrict__ f1,
                            const _Float16* __restrict__ f2w,
                            _Float16* __restrict__ corr)   // NHWC, 64 ch
{
    __shared__ _Float16 s1[CF];
    int pix = blockIdx.x;
    int x = pix % WW;
    int y = (pix / WW) % HH;
    int b = pix / (WW * HH);
    const _Float16* p1 = f1 + ((size_t)(b * HH + y) * WW + x) * CF;
    for (int i = threadIdx.x; i < CF; i += 64) s1[i] = p1[i];
    __syncthreads();
    int d = threadIdx.x;
    _Float16 res = (_Float16)0.f;
    if (d < 49) {
        int dy = d / 7 - 3, dx = d % 7 - 3;
        int ys = y + dy, xs = x + dx;
        float acc = 0.f;
        if (ys >= 0 && ys < HH && xs >= 0 && xs < WW) {
            const _Float16* p2 = f2w + ((size_t)(b * HH + ys) * WW + xs) * CF;
            #pragma unroll 8
            for (int c = 0; c < CF; ++c)
                acc += (float)s1[c] * (float)p2[c];
        }
        acc *= (1.0f / 96.0f);
        acc = (acc >= 0.f) ? acc : 0.1f * acc;
        res = (_Float16)acc;
    }
    corr[((size_t)(b * HH + y) * WW + x) * 64 + d] = res;
}

// ---------------------------------------------------------------------------
// 5) weight prep: OIHW f32 -> packed per-lane WMMA B fragments (f16)
//    layout: [((ky*KW+kx)*nslices + ks)*ntiles + ntile][lane][16 halves]
//    B 32x16 fragment as packed here (contiguous k, not the ISA table order): lane -> col N = lane%16, half = lane/16;
//    element e -> K = half*16 + e ; cin = ks*32 + K (zero if cin >= CIN)
// ---------------------------------------------------------------------------
__global__ void wprep_kernel(const float* __restrict__ w,   // (COUT,CIN,KH,KW)
                             _Float16* __restrict__ wpack,
                             int COUT, int CIN, int CINP, int KH, int KW)
{
    int ntiles = COUT / 16;
    int nslices = CINP / 32;
    int total = KH * KW * nslices * ntiles * 32;
    int idx = blockIdx.x * blockDim.x + threadIdx.x;
    if (idx >= total) return;
    int lane = idx & 31;
    int t = idx >> 5;
    int ntile = t % ntiles; t /= ntiles;
    int ks = t % nslices;  t /= nslices;
    int kx = t % KW;       t /= KW;
    int ky = t;
    int n = lane & 15, half = lane >> 4;
    int cout = ntile * 16 + n;
    _Float16* dst = wpack + (size_t)idx * 16;
    #pragma unroll
    for (int e = 0; e < 16; ++e) {
        int K = half * 16 + e;
        int cin = ks * 32 + K;
        float v = 0.f;
        if (cin < CIN)
            v = w[(((size_t)cout * CIN + cin) * KH + ky) * KW + kx];
        dst[e] = (_Float16)v;
    }
}

// ---------------------------------------------------------------------------
// 6) implicit-GEMM conv via V_WMMA_F32_16X16X32_F16
//    input NHWC f16 (Cin = NSLICES*32), output NHWC f16 (+bias, leaky)
//    block = 32*(COUT/16) threads; each wave: (16*MTILES) pixels x 16 outch,
//    MTILES independent accumulators (no WMMA->WMMA RAW hazard chains),
//    B fragment loaded once per k-slice and reused by MTILES WMMAs.
//    A 16x32 f16 order used here (contiguous k, same as B packing): lane -> row M = lane%16, half = lane/16;
//      elems 0..7 : K = half*16 + 0..7 ; elems 8..15 : K = half*16 + 8 + 0..7
// ---------------------------------------------------------------------------
template<int KH, int KW, int PAD, int NSLICES, int COUT, int MTILES>
__global__ void conv_wmma_kernel(const _Float16* __restrict__ in,
                                 const _Float16* __restrict__ wpack,
                                 const float* __restrict__ bias,
                                 _Float16* __restrict__ out)
{
    const int lane  = threadIdx.x & 31;
    const int ntile = threadIdx.x >> 5;
    const int ntiles = COUT / 16;
    const int tilesx = WW / (16 * MTILES);
    const int CIN = NSLICES * 32;

    int gid = blockIdx.x;
    int x0 = (gid % tilesx) * (16 * MTILES);
    int y  = (gid / tilesx) % HH;
    int b  = gid / (tilesx * HH);

    const int half = lane >> 4;
    const int m    = lane & 15;

    v8f acc[MTILES] = {};
    #pragma unroll
    for (int ky = 0; ky < KH; ++ky) {
        int ys = y + ky - PAD;
        if (ys < 0 || ys >= HH) continue;
        const _Float16* inrow = in + ((size_t)((b * HH + ys) * WW)) * CIN;
        #pragma unroll
        for (int kx = 0; kx < KW; ++kx) {
            #pragma unroll
            for (int ks = 0; ks < NSLICES; ++ks) {
                // B fragment: once per slice, reused by MTILES WMMAs
                const _Float16* wp = wpack +
                    ((size_t)(((ky * KW + kx) * NSLICES + ks) * ntiles + ntile)) * 512 +
                    lane * 16;
                v8h blo = *(const v8h*)(wp);
                v8h bhi = *(const v8h*)(wp + 8);
                v16h bf = __builtin_shufflevector(blo, bhi,
                    0,1,2,3,4,5,6,7,8,9,10,11,12,13,14,15);
                #pragma unroll
                for (int t = 0; t < MTILES; ++t) {
                    int xs = x0 + t * 16 + m + kx - PAD;
                    bool inb = (xs >= 0) && (xs < WW);
                    v8h alo = {}, ahi = {};
                    if (inb) {
                        const _Float16* arow = inrow + (size_t)xs * CIN;
                        alo = *(const v8h*)(arow + ks * 32 + half * 16);
                        ahi = *(const v8h*)(arow + ks * 32 + half * 16 + 8);
                    }
                    v16h a = __builtin_shufflevector(alo, ahi,
                        0,1,2,3,4,5,6,7,8,9,10,11,12,13,14,15);
                    acc[t] = __builtin_amdgcn_wmma_f32_16x16x32_f16(
                        false, a, false, bf, (short)0, acc[t], false, false);
                }
            }
        }
    }

    // C/D layout: lane -> col N (= outch), vgpr i -> row M = i + 8*half (= pixel)
    int cout = ntile * 16 + m;
    float bv = bias[cout];
    #pragma unroll
    for (int t = 0; t < MTILES; ++t) {
        _Float16* obase = out +
            ((size_t)((b * HH + y) * WW + x0 + t * 16)) * COUT + cout;
        #pragma unroll
        for (int i = 0; i < 8; ++i) {
            int mm = i + 8 * half;
            float v = acc[t][i] + bv;
            v = (v >= 0.f) ? v : 0.1f * v;
            obase[(size_t)mm * COUT] = (_Float16)v;
        }
    }
}

// ---------------------------------------------------------------------------
// 7) conv4 (32->2, 5x5, pad 2, no activation) + residual flow add, f32 NCHW out
// ---------------------------------------------------------------------------
__global__ void conv4_kernel(const _Float16* __restrict__ xin,  // NHWC 32ch
                             const float* __restrict__ w4,      // (2,32,5,5)
                             const float* __restrict__ b4,
                             const float* __restrict__ flow,    // NCHW f32
                             float* __restrict__ out)           // NCHW f32
{
    int idx = blockIdx.x * blockDim.x + threadIdx.x;
    const int n = BN * HH * WW;
    if (idx >= n) return;
    int x = idx % WW;
    int y = (idx / WW) % HH;
    int b = idx / (WW * HH);
    float a0 = b4[0], a1 = b4[1];
    for (int ky = 0; ky < 5; ++ky) {
        int ys = y + ky - 2;
        if (ys < 0 || ys >= HH) continue;
        for (int kx = 0; kx < 5; ++kx) {
            int xs = x + kx - 2;
            if (xs < 0 || xs >= WW) continue;
            const _Float16* p = xin + ((size_t)(b * HH + ys) * WW + xs) * 32;
            #pragma unroll
            for (int c = 0; c < 32; ++c) {
                float v = (float)p[c];
                a0 += v * w4[((0 * 32 + c) * 5 + ky) * 5 + kx];
                a1 += v * w4[((1 * 32 + c) * 5 + ky) * 5 + kx];
            }
        }
    }
    size_t o0 = ((size_t)(b * 2 + 0) * HH + y) * WW + x;
    size_t o1 = ((size_t)(b * 2 + 1) * HH + y) * WW + x;
    out[o0] = flow[o0] + a0;
    out[o1] = flow[o1] + a1;
}

// ---------------------------------------------------------------------------
// workspace layout (bytes, all 256-aligned)
// ---------------------------------------------------------------------------
#define OFF_FLOW  ((size_t)0)            //  1,179,648  f32 (B,2,192,192)
#define OFF_F1    ((size_t)1179648)      // 28,311,552  f16 NHWC
#define OFF_F2    ((size_t)29491200)     // 28,311,552  f16 NHWC
#define OFF_F2W   ((size_t)57802752)     // 28,311,552  f16 NHWC (warped)
#define OFF_CORR  ((size_t)86114304)     // 18,874,368  f16 NHWC 64ch
#define OFF_X1    ((size_t)104988672)    // 37,748,736  f16 NHWC 128ch
#define OFF_X2    ((size_t)142737408)    // 18,874,368  f16 NHWC 64ch
#define OFF_X3    ((size_t)161611776)    //  9,437,184  f16 NHWC 32ch
#define OFF_WP1   ((size_t)171048960)    //    147,456  f16 packed
#define OFF_WP2   ((size_t)171196416)    //    147,456  f16 packed
#define OFF_WP3   ((size_t)171343872)    //     36,864  f16 packed

extern "C" void kernel_launch(void* const* d_in, const int* in_sizes, int n_in,
                              void* d_out, int out_size, void* d_ws, size_t ws_size,
                              hipStream_t stream)
{
    // inputs (setup_inputs order); tensor1/tensor2 (d_in[0..1]) are unused
    const float* f1_nchw = (const float*)d_in[2];
    const float* f2_nchw = (const float*)d_in[3];
    const float* flow_in = (const float*)d_in[4];
    const float* w_up    = (const float*)d_in[5];
    const float* w1 = (const float*)d_in[6];
    const float* b1 = (const float*)d_in[7];
    const float* w2 = (const float*)d_in[8];
    const float* b2 = (const float*)d_in[9];
    const float* w3 = (const float*)d_in[10];
    const float* b3 = (const float*)d_in[11];
    const float* w4 = (const float*)d_in[12];
    const float* b4 = (const float*)d_in[13];
    float* out = (float*)d_out;

    char* ws = (char*)d_ws;
    float*    flow = (float*)   (ws + OFF_FLOW);
    _Float16* f1   = (_Float16*)(ws + OFF_F1);
    _Float16* f2   = (_Float16*)(ws + OFF_F2);
    _Float16* f2w  = (_Float16*)(ws + OFF_F2W);
    _Float16* corr = (_Float16*)(ws + OFF_CORR);
    _Float16* x1   = (_Float16*)(ws + OFF_X1);
    _Float16* x2   = (_Float16*)(ws + OFF_X2);
    _Float16* x3   = (_Float16*)(ws + OFF_X3);
    _Float16* wp1  = (_Float16*)(ws + OFF_WP1);
    _Float16* wp2  = (_Float16*)(ws + OFF_WP2);
    _Float16* wp3  = (_Float16*)(ws + OFF_WP3);

    const int npix = BN * HH * WW;                 // 147456

    // 1) upflow
    {
        int n = BN * 2 * HH * WW;
        upflow_kernel<<<(n + 255) / 256, 256, 0, stream>>>(flow_in, w_up, flow);
    }
    // 2) features -> NHWC f16
    {
        size_t n = (size_t)BN * CF * HH * WW;
        int blocks = (int)((n + 255) / 256);
        nchw_to_nhwc_f16<<<blocks, 256, 0, stream>>>(f1_nchw, f1);
        nchw_to_nhwc_f16<<<blocks, 256, 0, stream>>>(f2_nchw, f2);
    }
    // 3) weight packing (small)
    wprep_kernel<<<(9 * 2 * 8 * 32 + 255) / 256, 256, 0, stream>>>(w1, wp1, 128, 49, 64, 3, 3);
    wprep_kernel<<<(9 * 4 * 4 * 32 + 255) / 256, 256, 0, stream>>>(w2, wp2, 64, 128, 128, 3, 3);
    wprep_kernel<<<(9 * 2 * 2 * 32 + 255) / 256, 256, 0, stream>>>(w3, wp3, 32, 64, 64, 3, 3);
    // 4) backwarp (flow * 2.5)
    backwarp_kernel<<<npix, CF, 0, stream>>>(f2, flow, f2w);
    // 5) correlation + leaky -> 64ch NHWC f16
    corr_kernel<<<npix, 64, 0, stream>>>(f1, f2w, corr);
    // 6) WMMA convs (leaky fused), 64-pixel x-tiles (WW/64 = 3 per row)
    {
        int blocks = BN * HH * (WW / 64);          // 2304
        conv_wmma_kernel<3, 3, 1, 2, 128, 4><<<blocks, 256, 0, stream>>>(corr, wp1, b1, x1);
        conv_wmma_kernel<3, 3, 1, 4, 64,  4><<<blocks, 128, 0, stream>>>(x1, wp2, b2, x2);
        conv_wmma_kernel<3, 3, 1, 2, 32,  4><<<blocks, 64,  0, stream>>>(x2, wp3, b3, x3);
    }
    // 7) conv4 + flow residual
    conv4_kernel<<<(npix + 255) / 256, 256, 0, stream>>>(x3, w4, b4, flow, out);
}
